// DPLSTMLayer_88751204204958
// MI455X (gfx1250) — matrix operand order repaired and checked by reading, not compiled, not run on hardware
//
#include <hip/hip_runtime.h>

typedef __bf16 bf16;
typedef bf16  v16bf __attribute__((ext_vector_type(16)));
typedef bf16  v8bf  __attribute__((ext_vector_type(8)));
typedef float v8f   __attribute__((ext_vector_type(8)));
typedef int   v4i   __attribute__((ext_vector_type(4)));

// Problem sizes (fixed by the reference)
constexpr int T  = 512;
constexpr int Bz = 64;
constexpr int D  = 1024;
constexpr int H  = 1024;
constexpr int G  = 4 * H;          // 4096 gate columns
constexpr int M  = T * Bz;         // 32768 rows of the big GEMM

// ---------------------------------------------------------------------------
// Async global->LDS staging (CDNA5 GLOBAL_LOAD_ASYNC_TO_LDS_B128, ASYNCcnt).
// Signature probe-confirmed: (v4i as1* src, v4i as3* dst, imm offset, imm cpol)
// ---------------------------------------------------------------------------
#if __has_builtin(__builtin_amdgcn_global_load_async_to_lds_b128)
#define HAVE_ASYNC_LDS 1
#else
#define HAVE_ASYNC_LDS 0
#endif

#if HAVE_ASYNC_LDS
#if __has_builtin(__builtin_amdgcn_s_wait_asynccnt)
#define WAIT_ASYNC(n) __builtin_amdgcn_s_wait_asynccnt(n)
#else
#define WAIT_ASYNC(n) asm volatile("s_wait_asynccnt " #n ::: "memory")
#endif
typedef __attribute__((address_space(1))) v4i g_v4i;
typedef __attribute__((address_space(3))) v4i l_v4i;
#else
#define WAIT_ASYNC(n)
#endif

__device__ __forceinline__ void stage16(bf16* dst_lds, const bf16* src_g) {
#if HAVE_ASYNC_LDS
  __builtin_amdgcn_global_load_async_to_lds_b128(
      (g_v4i*)src_g, (l_v4i*)dst_lds, /*offset=*/0, /*cpol=*/0);
#else
  *(v8bf*)dst_lds = *(const v8bf*)src_g;
#endif
}

__device__ __forceinline__ v16bf cat8(v8bf lo, v8bf hi) {
  return __builtin_shufflevector(lo, hi, 0, 1, 2, 3, 4, 5, 6, 7,
                                 8, 9, 10, 11, 12, 13, 14, 15);
}

#define WMMA_BF16(a, b, c) \
  __builtin_amdgcn_wmma_f32_16x16x32_bf16(false, (a), false, (b), (short)0, (c), false, false)

// ---------------------------------------------------------------------------
// Elementwise helpers
// ---------------------------------------------------------------------------
__global__ void cvt_f32_to_bf16(const float* __restrict__ src,
                                bf16* __restrict__ dst, int n) {
  int i = blockIdx.x * blockDim.x + threadIdx.x;
  int stride = gridDim.x * blockDim.x;
  for (; i < n; i += stride) dst[i] = (bf16)src[i];
}

__global__ void copy_f32(const float* __restrict__ src,
                         float* __restrict__ dst, int n) {
  int i = blockIdx.x * blockDim.x + threadIdx.x;
  int stride = gridDim.x * blockDim.x;
  for (; i < n; i += stride) dst[i] = src[i];
}

// ---------------------------------------------------------------------------
// Phase 1: gates_x[m, g] = sum_d x[m, d] * w_ih[g, d] + b_ih[g]
//   Block: 256 threads = 8 waves; block tile 256 rows x 64 cols.
//   Each wave: 2 M-tiles x 4 N-tiles. w_ih k-chunks async-staged to LDS
//   (static double buffer, pipeline unrolled by 2), shared by all 8 waves.
//   Grid: (M/256, G/64) = (128, 64).
// ---------------------------------------------------------------------------
__global__ __launch_bounds__(256)
void gemm_gates_x(const bf16* __restrict__ xb,    // [M, D] bf16
                  const bf16* __restrict__ wih,   // [G, D] bf16
                  const float* __restrict__ bih,  // [G]
                  float* __restrict__ gx)         // [M, G] f32
{
  __shared__ alignas(16) bf16 wstage[2][64][40];  // rows padded 32->40 (bank-safe)

  const int tid  = threadIdx.x;
  const int lane = tid & 31;
  const int wave = tid >> 5;
  const int tb   = (blockIdx.x * 8 + wave) * 2;   // first of 2 m-tiles
  const int n0   = blockIdx.y * 64;
  const int col  = lane & 15;
  const int akb  = (lane >> 4) * 8;               // A fragment k-base
  const int bkb  = (lane >> 4) * 16;              // unused; B fragment also uses akb

  bf16 (* const sb0)[40] = wstage[0];
  bf16 (* const sb1)[40] = wstage[1];

  // Stage one 64x32 k-chunk of w_ih: 256 b128 transfers, one per thread.
  const int sr = tid >> 2, sseg = (tid & 3) * 8;
  auto stage = [&](bf16 (*buf)[40], int kc) {
    stage16(&buf[sr][sseg], wih + (size_t)(n0 + sr) * D + kc * 32 + sseg);
  };

  v8f acc[2][4] = {};
  const bf16* __restrict__ a0 = xb + (size_t)((tb + 0) * 16 + col) * D;
  const bf16* __restrict__ a1 = xb + (size_t)((tb + 1) * 16 + col) * D;

  auto compute = [&](const bf16 (*wb)[40], int k0) {
    v16bf a[2];
    a[0] = cat8(*(const v8bf*)(a0 + k0 + akb), *(const v8bf*)(a0 + k0 + 16 + akb));
    a[1] = cat8(*(const v8bf*)(a1 + k0 + akb), *(const v8bf*)(a1 + k0 + 16 + akb));
#pragma unroll
    for (int t = 0; t < 4; ++t) {
      v16bf b = cat8(*(const v8bf*)&wb[16 * t + col][akb], *(const v8bf*)&wb[16 * t + col][16 + akb]);
      acc[0][t] = WMMA_BF16(a[0], b, acc[0][t]);
      acc[1][t] = WMMA_BF16(a[1], b, acc[1][t]);
    }
  };

  constexpr int NC = D / 32;   // 32 chunks (even)
  stage(sb0, 0);
#pragma unroll 1
  for (int kc = 0; kc < NC; kc += 2) {
    stage(sb1, kc + 1);
    WAIT_ASYNC(1);             // chunk kc resident; kc+1 in flight
    __syncthreads();
    compute(sb0, kc * 32);
    __syncthreads();           // all waves done reading sb0
    if (kc + 2 < NC) { stage(sb0, kc + 2); WAIT_ASYNC(1); }
    else             { WAIT_ASYNC(0); }
    __syncthreads();
    compute(sb1, (kc + 1) * 32);
    __syncthreads();           // all waves done reading sb1
  }

  // C/D layout: element v of lane L -> row = tile*16 + v + 8*(L>>4), col = L&15
#pragma unroll
  for (int mi = 0; mi < 2; ++mi) {
    const int mbase = (tb + mi) * 16 + 8 * (lane >> 4);
#pragma unroll
    for (int t = 0; t < 4; ++t) {
      const int n = n0 + 16 * t + col;
      const float bias = bih[n];
#pragma unroll
      for (int v = 0; v < 8; ++v)
        gx[(size_t)(mbase + v) * G + n] = acc[mi][t][v] + bias;
    }
  }
}

// ---------------------------------------------------------------------------
// Phase 2: one timestep. Block = 128 threads = 4 gate-waves; one block per
//   16-wide hidden tile (grid = H/16 = 64). Wave g computes gate g for ALL 64
//   batch rows against one B fragment per k-chunk (w_hh read once per step).
//   h_prev k-chunks async-staged to LDS (static double buffer), shared by the
//   4 waves. Cell math block-local via LDS.
// ---------------------------------------------------------------------------
__global__ __launch_bounds__(128)
void lstm_step(const bf16* __restrict__ hprev,   // [Bz, H] bf16
               bf16* __restrict__ hnext,         // [Bz, H] bf16
               float* __restrict__ cbuf,         // [Bz, H] f32 in/out (tile-owned)
               const bf16* __restrict__ whh,     // [G, H] bf16
               const float* __restrict__ bhh,    // [G]
               const float* __restrict__ gx_t,   // [Bz, G] f32 slice
               float* __restrict__ hs_t)         // [Bz, H] f32 out slice
{
  __shared__ alignas(16) bf16 hstage[2][64][40]; // rows padded 32->40
  __shared__ float gl[4][64][16];

  const int tid  = threadIdx.x;
  const int lane = tid & 31;
  const int gate = tid >> 5;                     // 0=i, 1=f, 2=g, 3=o
  const int h0   = blockIdx.x * 16;
  const int col  = lane & 15;
  const int akb  = (lane >> 4) * 8;
  const int bkb  = (lane >> 4) * 16;

  bf16 (* const sb0)[40] = hstage[0];
  bf16 (* const sb1)[40] = hstage[1];

  // Stage one 64x32 k-chunk of h_prev: 256 b128 transfers, 2 per thread.
  const int sr0 = tid >> 2,         sseg0 = (tid & 3) * 8;
  const int sr1 = (tid + 128) >> 2, sseg1 = sseg0;
  auto stage = [&](bf16 (*buf)[40], int kc) {
    stage16(&buf[sr0][sseg0], hprev + (size_t)sr0 * H + kc * 32 + sseg0);
    stage16(&buf[sr1][sseg1], hprev + (size_t)sr1 * H + kc * 32 + sseg1);
  };

  v8f acc[4] = {};
  const bf16* __restrict__ wrow = whh + (size_t)(gate * H + h0 + col) * H;

  auto compute = [&](const bf16 (*hb)[40], int k0) {
    v16bf b = cat8(*(const v8bf*)(wrow + k0 + akb), *(const v8bf*)(wrow + k0 + 16 + akb));
#pragma unroll
    for (int bt = 0; bt < 4; ++bt) {
      v16bf a = cat8(*(const v8bf*)&hb[bt * 16 + col][akb],
                     *(const v8bf*)&hb[bt * 16 + col][16 + akb]);
      acc[bt] = WMMA_BF16(a, b, acc[bt]);
    }
  };

  constexpr int NC = H / 32;   // 32 chunks (even)
  stage(sb0, 0);
#pragma unroll 1
  for (int kc = 0; kc < NC; kc += 2) {
    stage(sb1, kc + 1);
    WAIT_ASYNC(2);             // chunk kc resident; kc+1 (2 issues/wave) in flight
    __syncthreads();
    compute(sb0, kc * 32);
    __syncthreads();
    if (kc + 2 < NC) { stage(sb0, kc + 2); WAIT_ASYNC(2); }
    else             { WAIT_ASYNC(0); }
    __syncthreads();
    compute(sb1, (kc + 1) * 32);
    __syncthreads();
  }

  // Add bias + precomputed input projection, stage gate pre-activations to LDS
  const int gcol = gate * H + h0 + col;
  const float bias = bhh[gcol];
  const int mb = 8 * (lane >> 4);
#pragma unroll
  for (int bt = 0; bt < 4; ++bt)
#pragma unroll
    for (int v = 0; v < 8; ++v) {
      const int r = bt * 16 + mb + v;
      gl[gate][r][col] = acc[bt][v] + bias + gx_t[(size_t)r * G + gcol];
    }
  __syncthreads();

  // Cell math: 64x16 elements, 128 threads -> 8 each
  for (int e = tid; e < 64 * 16; e += 128) {
    const int r  = e >> 4;
    const int cc = e & 15;
    const float i_in = gl[0][r][cc];
    const float f_in = gl[1][r][cc];
    const float g_in = gl[2][r][cc];
    const float o_in = gl[3][r][cc];
    const float it = 1.0f / (1.0f + __expf(-i_in));
    const float ft = 1.0f / (1.0f + __expf(-f_in));
    const float gt = tanhf(g_in);
    const float ot = 1.0f / (1.0f + __expf(-o_in));
    const size_t idx = (size_t)r * H + (h0 + cc);
    const float ct = ft * cbuf[idx] + it * gt;
    const float hv = ot * tanhf(ct);
    cbuf[idx]  = ct;
    hs_t[idx]  = hv;
    hnext[idx] = (bf16)hv;
  }
}

// ---------------------------------------------------------------------------
// Host-side orchestration
// ---------------------------------------------------------------------------
extern "C" void kernel_launch(void* const* d_in, const int* in_sizes, int n_in,
                              void* d_out, int out_size, void* d_ws, size_t ws_size,
                              hipStream_t stream) {
  (void)in_sizes; (void)n_in; (void)out_size; (void)ws_size;

  const float* x    = (const float*)d_in[0];   // [T, Bz, D]
  const float* h0   = (const float*)d_in[1];   // [Bz, H]
  const float* c0   = (const float*)d_in[2];   // [Bz, H]
  const float* w_ih = (const float*)d_in[3];   // [G, D]
  const float* b_ih = (const float*)d_in[4];   // [G]
  const float* w_hh = (const float*)d_in[5];   // [G, H]
  const float* b_hh = (const float*)d_in[6];   // [G]
  float* out = (float*)d_out;                  // hs [T*Bz*H] | h_n [Bz*H] | c_n [Bz*H]

  // Workspace layout
  char* ws = (char*)d_ws;
  float* gx    = (float*)ws;   ws += (size_t)M * G * sizeof(float);   // 512 MB
  bf16*  xb    = (bf16*)ws;    ws += (size_t)M * D * sizeof(bf16);    // 64 MB
  bf16*  wihb  = (bf16*)ws;    ws += (size_t)G * D * sizeof(bf16);    // 8 MB
  bf16*  whhb  = (bf16*)ws;    ws += (size_t)G * H * sizeof(bf16);    // 8 MB
  bf16*  hbuf0 = (bf16*)ws;    ws += (size_t)Bz * H * sizeof(bf16);
  bf16*  hbuf1 = (bf16*)ws;    ws += (size_t)Bz * H * sizeof(bf16);
  float* cbuf  = (float*)ws;   ws += (size_t)Bz * H * sizeof(float);

  auto cvt_grid = [](int n) { int g = (n + 255) / 256; return g > 16384 ? 16384 : g; };

  // Phase A: precision conversion / state init
  cvt_f32_to_bf16<<<cvt_grid(M * D), 256, 0, stream>>>(x,    xb,   M * D);
  cvt_f32_to_bf16<<<cvt_grid(G * D), 256, 0, stream>>>(w_ih, wihb, G * D);
  cvt_f32_to_bf16<<<cvt_grid(G * H), 256, 0, stream>>>(w_hh, whhb, G * H);
  cvt_f32_to_bf16<<<cvt_grid(Bz * H), 256, 0, stream>>>(h0,  hbuf0, Bz * H);
  copy_f32<<<cvt_grid(Bz * H), 256, 0, stream>>>(c0, cbuf, Bz * H);

  // Phase B: big input-projection GEMM (WMMA bf16 -> f32, LDS-staged B)
  {
    dim3 grid(M / 256, G / 64);   // (128, 64)
    gemm_gates_x<<<grid, 256, 0, stream>>>(xb, wihb, b_ih, gx);
  }

  // Phase C: 512 sequential steps; h double-buffered, c tile-owned in place
  for (int t = 0; t < T; ++t) {
    const bf16* hp = (t & 1) ? hbuf1 : hbuf0;
    bf16*       hn = (t & 1) ? hbuf0 : hbuf1;
    lstm_step<<<H / 16, 128, 0, stream>>>(
        hp, hn, cbuf, whhb, b_hh,
        gx + (size_t)t * Bz * G,
        out + (size_t)t * Bz * H);
  }

  // Final states: h_n = hs[T-1], c_n = cbuf
  float* hn_out = out + (size_t)T * Bz * H;
  float* cn_out = hn_out + (size_t)Bz * H;
  copy_f32<<<cvt_grid(Bz * H), 256, 0, stream>>>(out + (size_t)(T - 1) * Bz * H, hn_out, Bz * H);
  copy_f32<<<cvt_grid(Bz * H), 256, 0, stream>>>(cbuf, cn_out, Bz * H);
}
